// ToeplitzCausalLinear_77472620085401
// MI455X (gfx1250) — matrix operand order repaired and checked by reading, not compiled, not run on hardware
//
#include <hip/hip_runtime.h>

// ---- CDNA5 WMMA types ----
typedef __attribute__((ext_vector_type(16))) __bf16        v16bf;
typedef __attribute__((ext_vector_type(8)))  float         v8f;
typedef __attribute__((ext_vector_type(4)))  unsigned int  v4u;

#define TDIM 2048
#define EDIM 768
#define HD   256     // head dim = E/H
#define NB   8       // batch
#define BM   128
#define BN   64
#define BK   32

__device__ __forceinline__ unsigned short f2bf(float f) {
    // round-to-nearest-even f32 -> bf16
    unsigned int u = __float_as_uint(f);
    unsigned int r = u + 0x7FFFu + ((u >> 16) & 1u);
    return (unsigned short)(r >> 16);
}

// ---------------- one-pass f32 -> bf16 conversion of x into workspace ----------------
__global__ __launch_bounds__(256)
void cvt_x_bf16(const float* __restrict__ x, unsigned short* __restrict__ xb) {
    const int i = (blockIdx.x * 256 + threadIdx.x) * 4;   // total elems divisible by 1024
    const float4 v = *(const float4*)(x + i);
    unsigned int p0 = ((unsigned int)f2bf(v.y) << 16) | (unsigned int)f2bf(v.x);
    unsigned int p1 = ((unsigned int)f2bf(v.w) << 16) | (unsigned int)f2bf(v.z);
    uint2* dst = (uint2*)(xb + i);
    *dst = make_uint2(p0, p1);
}

// ---------------- Toeplitz-causal GEMM: Out_h = X_h * M_h  (M generated from w) ------
template <bool PRECONV>
__global__ __launch_bounds__(256)
void toeplitz_gemm(const void* __restrict__ xsrc,
                   const float* __restrict__ w,
                   const float* __restrict__ bias,
                   float* __restrict__ out)
{
    __shared__ __align__(16) unsigned short As[BM * BK];   // 8 KB  A tile (bf16, K contiguous)
    __shared__ __align__(16) unsigned short Bs[BN * BK];   // 4 KB  B tile (bf16, K contiguous per n)
    __shared__ __align__(16) unsigned short wlds[TDIM];    // 4 KB  head's Toeplitz vector (bf16)

    const int h    = blockIdx.z;
    const int m0   = blockIdx.y * BM;
    const int n0   = blockIdx.x * BN;
    const int tid  = threadIdx.x;
    const int lane = tid & 31;
    const int wv   = tid >> 5;
    const int wm   = (wv & 3) * 32;
    const int wn   = (wv >> 2) * 32;
    const int half = lane >> 4;
    const int l16  = lane & 15;

    // stage w[h] into LDS as bf16 (consumed after the first loop barrier)
    for (int i = tid; i < TDIM; i += 256)
        wlds[i] = f2bf(w[h * TDIM + i]);

    v8f acc[2][2];
    #pragma unroll
    for (int a = 0; a < 2; ++a)
        #pragma unroll
        for (int b = 0; b < 2; ++b)
            acc[a][b] = (v8f){0.f, 0.f, 0.f, 0.f, 0.f, 0.f, 0.f, 0.f};

    const int kTiles = (n0 + BN) / BK;   // causal upper bound
    const int kFull  = n0 / 32;          // kt < kFull  =>  whole tile strictly below diagonal band

    for (int kt = 0; kt < kTiles; ++kt) {
        const int k0 = kt * BK;
        __syncthreads();

        // ---- stage A tile (128 x 32) into LDS ----
        if (PRECONV) {
            const unsigned short* xb = (const unsigned short*)xsrc;
            #pragma unroll
            for (int it = 0; it < 2; ++it) {
                const int idx = tid + it * 256;          // 512 chunks of 8 ushorts
                const int row = idx >> 2;
                const int c8  = (idx & 3) * 8;
                const int rg  = m0 + row;
                const int bb  = rg >> 8;
                const int dd  = rg & 255;
                const unsigned short* src =
                    xb + ((size_t)(bb * EDIM + h * HD + dd)) * TDIM + k0 + c8;
                if (it == 0 && kt + 1 < kTiles)
                    __builtin_prefetch(src + BK, 0, 3);
                *(v4u*)&As[row * BK + c8] = *(const v4u*)src;
            }
        } else {
            const float* xf = (const float*)xsrc;
            #pragma unroll
            for (int it = 0; it < 4; ++it) {
                const int idx = tid + it * 256;          // 1024 float4 slots
                const int row = idx >> 3;
                const int c4  = (idx & 7) * 4;
                const int rg  = m0 + row;
                const int bb  = rg >> 8;
                const int dd  = rg & 255;
                const float* src = xf + ((size_t)(bb * EDIM + h * HD + dd)) * TDIM + k0 + c4;
                const float4 v = *(const float4*)src;
                if (it == 0 && kt + 1 < kTiles)
                    __builtin_prefetch(src + BK, 0, 3);
                unsigned int p0 = ((unsigned int)f2bf(v.y) << 16) | (unsigned int)f2bf(v.x);
                unsigned int p1 = ((unsigned int)f2bf(v.w) << 16) | (unsigned int)f2bf(v.z);
                unsigned int* dst = (unsigned int*)&As[row * BK + c4];
                dst[0] = p0; dst[1] = p1;
            }
        }

        // ---- cooperatively build B tile from the Toeplitz vector ----
        // Bs[n][k] = (n_g >= k_g) ? w[n_g - k_g] : 0 ; stored K-contiguous per column n
        const bool full = (kt < kFull);   // uniform: masking impossible
        #pragma unroll
        for (int it = 0; it < 4; ++it) {
            const int idx = tid + it * 256;              // 1024 dword slots
            const int n   = idx >> 4;                    // 0..63
            const int kp  = (idx & 15) * 2;              // even K
            const int e0  = (n0 + n) - (k0 + kp);
            unsigned int pv;
            if (full) {
                pv = ((unsigned int)wlds[e0 - 1] << 16) | (unsigned int)wlds[e0];
            } else {
                unsigned int lo = (e0 >= 1) ? (unsigned int)wlds[e0]     : (e0 == 0 ? (unsigned int)wlds[0] : 0u);
                unsigned int hi = (e0 >= 1) ? (unsigned int)wlds[e0 - 1] : 0u;
                pv = (hi << 16) | lo;
            }
            *(unsigned int*)&Bs[n * BK + kp] = pv;
        }
        __syncthreads();

        // ---- A fragments: two ds_load_b128 per 16x32 tile ----
        union { v16bf bf; v4u u4[2]; } afrag[2];
        #pragma unroll
        for (int a = 0; a < 2; ++a) {
            const unsigned short* base = &As[(wm + a * 16 + l16) * BK + half * 8];
            afrag[a].u4[0] = *(const v4u*)(base);
            afrag[a].u4[1] = *(const v4u*)(base + 16);
        }

        // ---- B fragments: two ds_load_b128 per 32x16 tile ----
        // lane holds N = l16; half h holds K 8h..8h+7 and 16+8h..16+8h+7 (same order as A)
        union { v16bf bf; v4u u4[2]; } bfrag[2];
        #pragma unroll
        for (int b = 0; b < 2; ++b) {
            const unsigned short* base = &Bs[(wn + b * 16 + l16) * BK + half * 8];
            bfrag[b].u4[0] = *(const v4u*)(base);
            bfrag[b].u4[1] = *(const v4u*)(base + 16);
        }

        // ---- 4 WMMAs per K step ----
        #pragma unroll
        for (int a = 0; a < 2; ++a)
            #pragma unroll
            for (int b = 0; b < 2; ++b)
                acc[a][b] = __builtin_amdgcn_wmma_f32_16x16x32_bf16(
                    false, afrag[a].bf, false, bfrag[b].bf,
                    (short)0, acc[a][b], false, false);
    }

    // ---- epilogue: C layout (VGPR r -> M = r + 8*half, N = lane%16), + bias ----
    #pragma unroll
    for (int b = 0; b < 2; ++b) {
        const int n  = n0 + wn + b * 16 + l16;
        const float bv = bias[h * TDIM + n];
        #pragma unroll
        for (int a = 0; a < 2; ++a) {
            #pragma unroll
            for (int r = 0; r < 8; ++r) {
                const int mrow = m0 + wm + a * 16 + half * 8 + r;
                const int bb   = mrow >> 8;
                const int dd   = mrow & 255;
                out[((size_t)(bb * EDIM + h * HD + dd)) * TDIM + n] = acc[a][b][r] + bv;
            }
        }
    }
}

extern "C" void kernel_launch(void* const* d_in, const int* in_sizes, int n_in,
                              void* d_out, int out_size, void* d_ws, size_t ws_size,
                              hipStream_t stream) {
    const float* x    = (const float*)d_in[0];   // (B, E, T)
    const float* wgt  = (const float*)d_in[1];   // (H, T)
    const float* bias = (const float*)d_in[2];   // (H, T)
    float* out        = (float*)d_out;           // (B, E, T)

    const size_t nx   = (size_t)NB * EDIM * TDIM;       // 12.58M elements
    const size_t need = nx * sizeof(unsigned short);    // 24 MB bf16 copy of x

    dim3 grid(TDIM / BN, (NB * HD) / BM, 3);            // (32, 16, 3)
    dim3 block(256);

    if (ws_size >= need) {
        unsigned short* xb = (unsigned short*)d_ws;
        cvt_x_bf16<<<dim3((unsigned)(nx / 4 / 256)), block, 0, stream>>>(x, xb);
        toeplitz_gemm<true><<<grid, block, 0, stream>>>(xb, wgt, bias, out);
    } else {
        toeplitz_gemm<false><<<grid, block, 0, stream>>>(x, wgt, bias, out);
    }
}
